// RQVAE_86397562126690
// MI455X (gfx1250) — compile-verified, matrix operand order checked by reading, not run on hardware
//
#include <hip/hip_runtime.h>
#include <hip/hip_bf16.h>

// ---------------------------------------------------------------------------
// RQ-VAE forward for MI455X (gfx1250, wave32).
// GEMMs: v_wmma_f32_16x16x32_bf16, LDS staged via Tensor Data Mover (TDM)
// double-buffered tensor_load_to_lds + s_wait_tensorcnt.
// ---------------------------------------------------------------------------

typedef __attribute__((ext_vector_type(16))) __bf16        bf16x16;
typedef __attribute__((ext_vector_type(8)))  float         f32x8;
typedef __attribute__((ext_vector_type(4)))  unsigned int  u32x4;
typedef __attribute__((ext_vector_type(8)))  int           i32x8;
typedef __attribute__((ext_vector_type(4)))  int           i32x4;

#if defined(__gfx1250__) && __has_builtin(__builtin_amdgcn_tensor_load_to_lds)
#define USE_TDM 1
#else
#define USE_TDM 0
#endif

// round-to-nearest-even f32 -> bf16 (ushort bits)
__device__ __forceinline__ unsigned short f2bf(float f) {
    unsigned u = __float_as_uint(f);
    u += 0x7FFFu + ((u >> 16) & 1u);
    return (unsigned short)(u >> 16);
}

// 16-element bf16 fragment (32B) from LDS via two aligned b128 loads
__device__ __forceinline__ bf16x16 ld_frag32(const unsigned short* p) {
    union { bf16x16 v; uint4 q[2]; } u;
    u.q[0] = *(const uint4*)(p);
    u.q[1] = *(const uint4*)(p + 8);
    return u.v;
}

#if USE_TDM
// Issue one TDM 2D tile load: tileRows x 32 bf16, row stride = strideElems,
// into LDS at lds_addr with 16B padding appended per 64B row (LDA=40 layout).
__device__ __forceinline__ void tdm_load_2d(unsigned lds_addr,
                                            const unsigned short* gptr,
                                            unsigned tileRows,
                                            unsigned strideElems) {
    unsigned long long ga = (unsigned long long)gptr;
    u32x4 g0;
    g0[0] = 1u;                                                // count=1 (valid user D#)
    g0[1] = lds_addr;                                          // LDS byte address
    g0[2] = (unsigned)ga;                                      // global addr [31:0]
    g0[3] = (unsigned)((ga >> 32) & 0x1FFFFFFull) | (2u << 30);// addr [56:32] | type=2
    i32x8 g1;
    // data_size=1 (2B) | pad_enable | pad_interval=3 (16 DW) | pad_amount=3 (4 DW)
    g1[0] = (int)((1u << 16) | (1u << 20) | (3u << 22) | (3u << 25));
    g1[1] = (int)(32u << 16);                    // tensor_dim0 = 32 (lo16 -> [31:16])
    g1[2] = (int)((tileRows & 0xFFFFu) << 16);   // dim0 hi=0 | tensor_dim1 lo16
    g1[3] = (int)(32u << 16);                    // dim1 hi=0 | tile_dim0 = 32
    g1[4] = (int)(tileRows & 0xFFFFu);           // tile_dim1 | tile_dim2 = 0
    g1[5] = (int)strideElems;                    // tensor_dim0_stride [31:0]
    g1[6] = 0;                                   // stride hi | dim1_stride lo
    g1[7] = 0;                                   // dim1_stride hi
    i32x4 gz  = {0, 0, 0, 0};
#if __clang_major__ >= 23
    i32x8 gz8 = {0, 0, 0, 0, 0, 0, 0, 0};
    __builtin_amdgcn_tensor_load_to_lds(g0, g1, gz, gz, gz8, 0);
#else
    __builtin_amdgcn_tensor_load_to_lds(g0, g1, gz, gz, 0);
#endif
}
#endif

// ---------------------------------------------------------------------------
// Prep kernels
// ---------------------------------------------------------------------------

__global__ __launch_bounds__(256) void f32_to_bf16_vec(
    const float4* __restrict__ in, ushort4* __restrict__ out, int n4) {
    int i = blockIdx.x * 256 + threadIdx.x;
    if (i < n4) {
        float4 v = in[i];
        ushort4 o;
        o.x = f2bf(v.x); o.y = f2bf(v.y); o.z = f2bf(v.z); o.w = f2bf(v.w);
        out[i] = o;
    }
}

__global__ __launch_bounds__(256) void transpose_to_bf16(
    const float* __restrict__ in, unsigned short* __restrict__ out,
    int rows, int cols) {
    long i = (long)blockIdx.x * 256 + threadIdx.x;
    if (i < (long)rows * cols) {
        int r = (int)(i / cols), c = (int)(i % cols);
        out[(long)c * rows + r] = f2bf(in[i]);
    }
}

__global__ __launch_bounds__(256) void code_norms(
    const float* __restrict__ emb, float* __restrict__ norms, int K, int C) {
    int k = blockIdx.x * 256 + threadIdx.x;
    if (k < K) {
        float s = 0.f;
        const float* p = emb + (long)k * C;
        for (int c = 0; c < C; ++c) { float v = p[c]; s += v * v; }
        norms[k] = s;
    }
}

__global__ __launch_bounds__(256) void make_dec_in(
    const float4* __restrict__ q1, const float4* __restrict__ q2,
    const float4* __restrict__ q3, ushort4* __restrict__ out, int n4) {
    int i = blockIdx.x * 256 + threadIdx.x;
    if (i < n4) {
        float4 a = q1[i], b = q2[i], c = q3[i];
        ushort4 o;
        o.x = f2bf(a.x + b.x + c.x);
        o.y = f2bf(a.y + b.y + c.y);
        o.z = f2bf(a.z + b.z + c.z);
        o.w = f2bf(a.w + b.w + c.w);
        out[i] = o;
    }
}

// ---------------------------------------------------------------------------
// WMMA GEMM:  C(MxN) = act(A(MxK,bf16) @ Bt(NxK,bf16)^T + bias)
// block tile 128x64, 8 waves, wave tile 32x32 (2x2 WMMA), K-step 32.
// LDS staging: double-buffered TDM tensor loads (wave 0 issues, TENSORcnt).
// ACT: 0=none, 1=relu, 2=sigmoid.  WF/WB select fp32 / bf16 outputs.
// ---------------------------------------------------------------------------

#define LDA 40  // 32 data + 8 pad bf16 elems -> 80B rows, conflict-free b128

template <int ACT, bool WF, bool WB>
__global__ __launch_bounds__(256) void gemm_bias_act(
    const unsigned short* __restrict__ A, const unsigned short* __restrict__ Bt,
    const float* __restrict__ bias,
    float* __restrict__ outF, unsigned short* __restrict__ outB,
    int M, int N, int K) {
    __shared__ alignas(16) unsigned short As[2][128 * LDA];
    __shared__ alignas(16) unsigned short Bs[2][64 * LDA];

    const int tid   = threadIdx.x;
    const int wave  = tid >> 5;
    const int lane  = tid & 31;
    const int waveM = wave >> 1;          // 0..3
    const int waveN = wave & 1;           // 0..1
    const int lm    = lane & 15;
    const int kh    = (lane >> 4) * 16;

    const long rowBase = (long)blockIdx.y * 128;
    const long colBase = (long)blockIdx.x * 64;
    const int  nk      = K >> 5;

    f32x8 acc[2][2] = {};

#if USE_TDM
    const unsigned ldsA0 = (unsigned)(unsigned long long)&As[0][0];
    const unsigned ldsA1 = (unsigned)(unsigned long long)&As[1][0];
    const unsigned ldsB0 = (unsigned)(unsigned long long)&Bs[0][0];
    const unsigned ldsB1 = (unsigned)(unsigned long long)&Bs[1][0];
    if (wave == 0) {   // prologue: tile 0 in flight
        tdm_load_2d(ldsA0, A + rowBase * K, 128, (unsigned)K);
        tdm_load_2d(ldsB0, Bt + colBase * K, 64, (unsigned)K);
    }
#else
    const int arow  = tid >> 1;           // 0..127
    const int ahalf = tid & 1;
    const int brow  = tid >> 2;           // 0..63
    const int bq    = tid & 3;
    const unsigned short* gA = A  + (rowBase + arow) * K + ahalf * 16;
    const unsigned short* gB = Bt + (colBase + brow) * K + bq * 8;
#endif

    for (int ks = 0; ks < nk; ++ks) {
        const int cur = ks & 1;
#if USE_TDM
        if (wave == 0) {
            if (ks + 1 < nk) {   // issue next tile, then wait for current one
                tdm_load_2d(cur ? ldsA0 : ldsA1,
                            A + rowBase * K + (long)(ks + 1) * 32, 128, (unsigned)K);
                tdm_load_2d(cur ? ldsB0 : ldsB1,
                            Bt + colBase * K + (long)(ks + 1) * 32, 64, (unsigned)K);
                __builtin_amdgcn_s_wait_tensorcnt(2);
            } else {
                __builtin_amdgcn_s_wait_tensorcnt(0);
            }
        }
        __syncthreads();
#else
        {
            const int k0 = ks * 32;
            const uint4* ga = (const uint4*)(gA + k0);
            uint4 a0 = ga[0];
            uint4 a1 = ga[1];
            uint4 b0 = *(const uint4*)(gB + k0);
            *(uint4*)&As[cur][arow * LDA + ahalf * 16]     = a0;
            *(uint4*)&As[cur][arow * LDA + ahalf * 16 + 8] = a1;
            *(uint4*)&Bs[cur][brow * LDA + bq * 8]         = b0;
            __syncthreads();
        }
#endif
        bf16x16 af[2], bfr[2];
#pragma unroll
        for (int i = 0; i < 2; ++i)
            af[i] = ld_frag32(&As[cur][(waveM * 32 + i * 16 + lm) * LDA + kh]);
#pragma unroll
        for (int j = 0; j < 2; ++j)
            bfr[j] = ld_frag32(&Bs[cur][(waveN * 32 + j * 16 + lm) * LDA + kh]);
#pragma unroll
        for (int i = 0; i < 2; ++i)
#pragma unroll
            for (int j = 0; j < 2; ++j)
                acc[i][j] = __builtin_amdgcn_wmma_f32_16x16x32_bf16(
                    false, af[i], false, bfr[j], (short)0, acc[i][j], false, false);
        __syncthreads();
    }

    // branch-free epilogue: bias + activation + store(s)
#pragma unroll
    for (int i = 0; i < 2; ++i) {
        const long mb = rowBase + waveM * 32 + i * 16 + (lane < 16 ? 0 : 8);
#pragma unroll
        for (int j = 0; j < 2; ++j) {
            const long nb = colBase + waveN * 32 + j * 16 + lm;
            const float bv = bias[nb];
#pragma unroll
            for (int r = 0; r < 8; ++r) {
                float v = acc[i][j][r] + bv;
                if constexpr (ACT == 1) v = v > 0.f ? v : 0.f;
                if constexpr (ACT == 2) v = __builtin_amdgcn_rcpf(1.f + __expf(-v));
                const long o = (mb + r) * N + nb;
                if constexpr (WF) outF[o] = v;
                if constexpr (WB) outB[o] = f2bf(v);
            }
        }
    }
}

// ---------------------------------------------------------------------------
// Fused residual-VQ stage: 64 rows/block, argmin_k (||e_k||^2 - 2 ze.e_k)
// over 1024 codes via WMMA score chunks (16 x 64 codes), then zq / ze_next.
// ---------------------------------------------------------------------------

#define LDQ 264  // 256 + 8 pad (bf16 elems)

__global__ __launch_bounds__(256) void rq_quantize(
    const unsigned short* __restrict__ zeB,  // N x 256 bf16
    const float* __restrict__ zeF,           // N x 256 f32
    const float* __restrict__ embF,          // 1024 x 256 f32
    const unsigned short* __restrict__ embB, // 1024 x 256 bf16
    const float* __restrict__ norms,         // 1024
    int* __restrict__ idxOut,                // N
    float* __restrict__ zqOut,               // N x 256
    float* __restrict__ zenF,                // N x 256 (nullable)
    unsigned short* __restrict__ zenB) {     // N x 256 bf16 (nullable)
    __shared__ alignas(16) unsigned short As[64 * LDQ];
    __shared__ alignas(16) unsigned short Bs[64 * LDQ];
    __shared__ float Ss[64 * 65];
    __shared__ float bestV[256];
    __shared__ int   bestI[256];
    __shared__ int   idxSh[64];

    const int tid  = threadIdx.x;
    const int wave = tid >> 5;
    const int lane = tid & 31;
    const int lm   = lane & 15;
    const int kh   = (lane >> 4) * 16;
    const long rowBase = (long)blockIdx.x * 64;

    const int lrow = tid >> 2;   // 0..63
    const int lseg = tid & 3;    // 0..3

    {   // stage ze tile 64x256 into LDS
        const uint4* g = (const uint4*)(zeB + (rowBase + lrow) * 256 + lseg * 64);
        uint4* s = (uint4*)&As[lrow * LDQ + lseg * 64];
#pragma unroll
        for (int j = 0; j < 8; ++j) s[j] = g[j];
    }
    float bv = 3.4e38f;
    int   bi = 0;
    __syncthreads();

    const int t0 = wave * 2;     // two 16x16 tiles per wave in the 4x4 grid
    const int tm = t0 >> 2;

    for (int c = 0; c < 16; ++c) {
        {   // stage 64 codes x 256 into LDS (+ prefetch next chunk)
            const uint4* g = (const uint4*)(embB + ((long)c * 64 + lrow) * 256 + lseg * 64);
            if (c + 1 < 16)
                __builtin_prefetch(
                    (const void*)(embB + ((long)(c + 1) * 64 + lrow) * 256 + lseg * 64), 0, 3);
            uint4* s = (uint4*)&Bs[lrow * LDQ + lseg * 64];
#pragma unroll
            for (int j = 0; j < 8; ++j) s[j] = g[j];
        }
        __syncthreads();

        f32x8 acc[2] = {};
#pragma unroll
        for (int ks = 0; ks < 8; ++ks) {
            bf16x16 a = ld_frag32(&As[(tm * 16 + lm) * LDQ + ks * 32 + kh]);
#pragma unroll
            for (int i = 0; i < 2; ++i) {
                const int tn = (t0 + i) & 3;
                bf16x16 b = ld_frag32(&Bs[(tn * 16 + lm) * LDQ + ks * 32 + kh]);
                acc[i] = __builtin_amdgcn_wmma_f32_16x16x32_bf16(
                    false, a, false, b, (short)0, acc[i], false, false);
            }
        }
#pragma unroll
        for (int i = 0; i < 2; ++i) {
            const int tn = (t0 + i) & 3;
            const int mb = tm * 16 + (lane < 16 ? 0 : 8);
            const int nb = tn * 16 + lm;
#pragma unroll
            for (int r = 0; r < 8; ++r) Ss[(mb + r) * 65 + nb] = acc[i][r];
        }
        __syncthreads();

        {   // running argmin of (||e||^2 - 2*score); ||ze||^2 row-constant
            const int row = tid & 63, part = tid >> 6;
#pragma unroll
            for (int j = 0; j < 16; ++j) {
                const int col  = part * 16 + j;
                const int gcol = c * 64 + col;
                const float v = norms[gcol] - 2.f * Ss[row * 65 + col];
                if (v < bv) { bv = v; bi = gcol; }
            }
        }
        __syncthreads();
    }

    bestV[(tid & 63) * 4 + (tid >> 6)] = bv;
    bestI[(tid & 63) * 4 + (tid >> 6)] = bi;
    __syncthreads();
    if (tid < 64) {
        float v = bestV[tid * 4];
        int  ix = bestI[tid * 4];
#pragma unroll
        for (int p = 1; p < 4; ++p) {
            float vp = bestV[tid * 4 + p];
            int   ip = bestI[tid * 4 + p];
            if (vp < v || (vp == v && ip < ix)) { v = vp; ix = ip; }
        }
        idxSh[tid] = ix;
        idxOut[rowBase + tid] = ix;
    }
    __syncthreads();

    {   // zq = emb[idx];  ze_next = ze - zq (fp32 + bf16)
        const long grow = rowBase + lrow;
        const int  kidx = idxSh[lrow];
        const long base = grow * 256 + lseg * 64;
        const float4* ep = (const float4*)(embF + (long)kidx * 256 + lseg * 64);
        const float4* zp = (const float4*)(zeF + base);
        float4* zq = (float4*)(zqOut + base);
#pragma unroll
        for (int j = 0; j < 16; ++j) {
            float4 e = ep[j];
            zq[j] = e;
            if (zenF) {
                float4 z = zp[j];
                float4 d;
                d.x = z.x - e.x; d.y = z.y - e.y; d.z = z.z - e.z; d.w = z.w - e.w;
                ((float4*)(zenF + base))[j] = d;
                ushort4 db;
                db.x = f2bf(d.x); db.y = f2bf(d.y); db.z = f2bf(d.z); db.w = f2bf(d.w);
                ((ushort4*)(zenB + base))[j] = db;
            }
        }
    }
}

// ---------------------------------------------------------------------------
// Host-side orchestration
// ---------------------------------------------------------------------------

extern "C" void kernel_launch(void* const* d_in, const int* in_sizes, int n_in,
                              void* d_out, int out_size, void* d_ws, size_t ws_size,
                              hipStream_t stream) {
    (void)in_sizes; (void)n_in; (void)out_size; (void)ws_size;

    const float* x    = (const float*)d_in[0];
    const float* We1  = (const float*)d_in[1];
    const float* be1  = (const float*)d_in[2];
    const float* We2  = (const float*)d_in[3];
    const float* be2  = (const float*)d_in[4];
    const float* Wd1  = (const float*)d_in[5];
    const float* bd1  = (const float*)d_in[6];
    const float* Wd2  = (const float*)d_in[7];
    const float* bd2  = (const float*)d_in[8];
    const float* emb1 = (const float*)d_in[9];
    const float* emb2 = (const float*)d_in[10];
    const float* emb3 = (const float*)d_in[11];

    constexpr long NB = 32768, IN = 768, HID = 2048, OD = 256, KC = 1024;

    float* out   = (float*)d_out;
    float* recon = out;                        // NB x IN
    float* ze1   = recon + NB * IN;            // NB x OD
    float* ze2   = ze1 + NB * OD;
    float* ze3   = ze2 + NB * OD;
    float* zq1   = ze3 + NB * OD;
    float* zq2   = zq1 + NB * OD;
    float* zq3   = zq2 + NB * OD;
    int*   n1    = (int*)(zq3 + NB * OD);      // NB ints
    int*   n2    = n1 + NB;
    int*   n3    = n2 + NB;

    char* ws = (char*)d_ws;
    size_t off = 0;
    auto alloc = [&](size_t bytes) {
        void* p = ws + off;
        off += (bytes + 255) & ~(size_t)255;
        return p;
    };
    unsigned short* xb   = (unsigned short*)alloc(NB * IN * 2);
    unsigned short* We1t = (unsigned short*)alloc(HID * IN * 2);
    unsigned short* We2t = (unsigned short*)alloc(OD * HID * 2);
    unsigned short* Wd1t = (unsigned short*)alloc(HID * OD * 2);
    unsigned short* Wd2t = (unsigned short*)alloc(IN * HID * 2);
    unsigned short* e1b  = (unsigned short*)alloc(KC * OD * 2);
    unsigned short* e2b  = (unsigned short*)alloc(KC * OD * 2);
    unsigned short* e3b  = (unsigned short*)alloc(KC * OD * 2);
    float* nrm1 = (float*)alloc(KC * 4);
    float* nrm2 = (float*)alloc(KC * 4);
    float* nrm3 = (float*)alloc(KC * 4);
    unsigned short* hb   = (unsigned short*)alloc(NB * HID * 2);  // reused in decoder
    unsigned short* z1b  = (unsigned short*)alloc(NB * OD * 2);
    unsigned short* z2b  = (unsigned short*)alloc(NB * OD * 2);
    unsigned short* z3b  = (unsigned short*)alloc(NB * OD * 2);
    unsigned short* decb = (unsigned short*)alloc(NB * OD * 2);

    // --- prep ---
    f32_to_bf16_vec<<<(int)(NB * IN / 4 / 256), 256, 0, stream>>>(
        (const float4*)x, (ushort4*)xb, (int)(NB * IN / 4));
    transpose_to_bf16<<<(int)((IN * HID + 255) / 256), 256, 0, stream>>>(We1, We1t, (int)IN, (int)HID);
    transpose_to_bf16<<<(int)((HID * OD + 255) / 256), 256, 0, stream>>>(We2, We2t, (int)HID, (int)OD);
    transpose_to_bf16<<<(int)((OD * HID + 255) / 256), 256, 0, stream>>>(Wd1, Wd1t, (int)OD, (int)HID);
    transpose_to_bf16<<<(int)((HID * IN + 255) / 256), 256, 0, stream>>>(Wd2, Wd2t, (int)HID, (int)IN);
    f32_to_bf16_vec<<<(int)(KC * OD / 4 / 256), 256, 0, stream>>>(
        (const float4*)emb1, (ushort4*)e1b, (int)(KC * OD / 4));
    f32_to_bf16_vec<<<(int)(KC * OD / 4 / 256), 256, 0, stream>>>(
        (const float4*)emb2, (ushort4*)e2b, (int)(KC * OD / 4));
    f32_to_bf16_vec<<<(int)(KC * OD / 4 / 256), 256, 0, stream>>>(
        (const float4*)emb3, (ushort4*)e3b, (int)(KC * OD / 4));
    code_norms<<<(int)(KC / 256), 256, 0, stream>>>(emb1, nrm1, (int)KC, (int)OD);
    code_norms<<<(int)(KC / 256), 256, 0, stream>>>(emb2, nrm2, (int)KC, (int)OD);
    code_norms<<<(int)(KC / 256), 256, 0, stream>>>(emb3, nrm3, (int)KC, (int)OD);

    // --- encoder ---
    gemm_bias_act<1, false, true><<<dim3((int)(HID / 64), (int)(NB / 128)), 256, 0, stream>>>(
        xb, We1t, be1, nullptr, hb, (int)NB, (int)HID, (int)IN);              // relu
    gemm_bias_act<0, true, true><<<dim3((int)(OD / 64), (int)(NB / 128)), 256, 0, stream>>>(
        hb, We2t, be2, ze1, z1b, (int)NB, (int)OD, (int)HID);                 // ze1

    // --- residual VQ ---
    rq_quantize<<<(int)(NB / 64), 256, 0, stream>>>(
        z1b, ze1, emb1, e1b, nrm1, n1, zq1, ze2, z2b);
    rq_quantize<<<(int)(NB / 64), 256, 0, stream>>>(
        z2b, ze2, emb2, e2b, nrm2, n2, zq2, ze3, z3b);
    rq_quantize<<<(int)(NB / 64), 256, 0, stream>>>(
        z3b, ze3, emb3, e3b, nrm3, n3, zq3, nullptr, nullptr);

    // --- straight-through decoder input ---
    make_dec_in<<<(int)(NB * OD / 4 / 256), 256, 0, stream>>>(
        (const float4*)zq1, (const float4*)zq2, (const float4*)zq3,
        (ushort4*)decb, (int)(NB * OD / 4));

    // --- decoder ---
    gemm_bias_act<2, false, true><<<dim3((int)(HID / 64), (int)(NB / 128)), 256, 0, stream>>>(
        decb, Wd1t, bd1, nullptr, hb, (int)NB, (int)HID, (int)OD);            // sigmoid
    gemm_bias_act<0, true, false><<<dim3((int)(IN / 64), (int)(NB / 128)), 256, 0, stream>>>(
        hb, Wd2t, bd2, recon, nullptr, (int)NB, (int)IN, (int)HID);           // recon
}
